// VQVAE_61838939128204
// MI455X (gfx1250) — matrix operand order repaired and checked by reading, compiled, not run on hardware
//
#include <hip/hip_runtime.h>
#include <hip/hip_bf16.h>
#include <stdint.h>

// ---------------------------------------------------------------------------
// CDNA5 (gfx1250) VQ-VAE forward. wave32, WMMA f32_16x16x32_f16 everywhere,
// 2x2 register-blocked implicit GEMM (32 cout x 32 pixels per wave).
// ---------------------------------------------------------------------------

typedef __attribute__((ext_vector_type(16))) _Float16 v16h;
typedef __attribute__((ext_vector_type(8)))  _Float16 v8h;
typedef __attribute__((ext_vector_type(8)))  float    v8f;

union FragU { v16h h; uint32_t u[8]; };

__device__ __forceinline__ v8f zero8() {
  v8f z = {0.f, 0.f, 0.f, 0.f, 0.f, 0.f, 0.f, 0.f};
  return z;
}

// A fragment (16x32 f16, M on lanes, K across regs, contiguous per half):
// lane L: m = L&15, half = L>>4 ; reg r: k = half*16 + 2r
__device__ __forceinline__ v16h load_a_frag(const _Float16* __restrict__ rowPtr,
                                            int half8) {
  FragU f;
#pragma unroll
  for (int r = 0; r < 8; ++r) {
    const int k = (half8 << 1) + (r << 1);
    f.u[r] = *reinterpret_cast<const uint32_t*>(rowPtr + k);
  }
  return f.h;
}

// B fragment (32x16 f16, N on lanes, K across regs, same contiguous order as A):
// lane L: n = L&15, half = L>>4 ; reg r: k = half*16 + 2r
// colPtr points at the K-contiguous run for this lane's column.
__device__ __forceinline__ v16h load_b_frag(const _Float16* __restrict__ colPtr,
                                            int half16, bool valid) {
  FragU f;
#pragma unroll
  for (int r = 0; r < 8; ++r) {
    const int k = half16 + (r << 1);
    f.u[r] = valid ? *reinterpret_cast<const uint32_t*>(colPtr + k) : 0u;
  }
  return f.h;
}

__device__ __forceinline__ v8f wmma16(v16h a, v16h b, v8f c) {
  return __builtin_amdgcn_wmma_f32_16x16x32_f16(false, a, false, b, (short)0,
                                                c, false, false);
}

// ---------------------------------------------------------------------------
// Weight repack: w[O][I][KH][KW] f32 -> wp[OPAD][KPAD] f16, K = tap*I + ci
// (tap = ky*KW + kx). Pads (o >= O or k >= KHW*I) are zero-filled.
// ---------------------------------------------------------------------------
__global__ void vq_pack_w(const float* __restrict__ w, _Float16* __restrict__ wp,
                          int O, int I, int KHW, int KPAD, int OPAD) {
  const long long t = (long long)blockIdx.x * blockDim.x + threadIdx.x;
  const long long total = (long long)OPAD * KPAD;
  if (t >= total) return;
  const int k = (int)(t % KPAD);
  const int o = (int)(t / KPAD);
  float v = 0.f;
  if (o < O && k < KHW * I) {
    const int tap = k / I;
    const int ci  = k % I;
    v = w[((long long)o * I + ci) * KHW + tap];
  }
  wp[t] = (_Float16)v;
}

// ---------------------------------------------------------------------------
// enc0 im2col: x f32 NCHW [NB,3,256,256] -> col f16 [NB*128*128][64]
// k = tap*3 + ci for k<48, zero-padded to 64.  (k4 s2 p1)
// ---------------------------------------------------------------------------
__global__ void vq_im2col_e0(const float* __restrict__ x,
                             _Float16* __restrict__ col, int NB) {
  const long long t = (long long)blockIdx.x * blockDim.x + threadIdx.x;
  const long long total = (long long)NB * 128 * 128 * 64;
  if (t >= total) return;
  const int k = (int)(t & 63);
  const long long pix = t >> 6;
  const int ox = (int)(pix & 127);
  const int oy = (int)((pix >> 7) & 127);
  const int nb = (int)(pix >> 14);
  float v = 0.f;
  if (k < 48) {
    const int tap = k / 3, ci = k % 3;
    const int ky = tap >> 2, kx = tap & 3;
    const int iy = oy * 2 - 1 + ky;
    const int ix = ox * 2 - 1 + kx;
    if (iy >= 0 && iy < 256 && ix >= 0 && ix < 256)
      v = x[(((long long)nb * 3 + ci) * 256 + iy) * 256 + ix];
  }
  col[t] = (_Float16)v;
}

// ---------------------------------------------------------------------------
// Forward conv as implicit GEMM, 2x2 register blocking.
// in: NHWC f16 [NB,IH,IW,CIN].  wp: [COUT][KH*KW*CIN], K=(ky*KW+kx)*CIN+ci.
// CIN multiple of 32.  One wave -> (16*MB couts) x (32 pixels).
// ---------------------------------------------------------------------------
template <int CIN, int COUT, int KH, int KW, int STRIDE, int PAD, bool RELU,
          bool WF16, bool WF32, int COUT_REAL>
__global__ __launch_bounds__(128) void vq_conv_fwd(
    const _Float16* __restrict__ in, const _Float16* __restrict__ wp,
    const float* __restrict__ bias, _Float16* __restrict__ out16,
    float* __restrict__ out32, int NB, int IH, int IW, int OH, int OW) {
  constexpr int KTOT = KH * KW * CIN;
  constexpr int MB = (COUT >= 32) ? 2 : 1;  // 16-row cout tiles per wave
  const int lane = threadIdx.x & 31;
  const int wave = threadIdx.x >> 5;
  const int n16 = lane & 15;
  const int half8 = (lane >> 4) << 3;
  const int half16 = (lane >> 4) << 4;

  const int npix = NB * OH * OW;
  const int pixTiles = (npix + 31) >> 5;      // 32 pixels per wave
  const int coutTiles = COUT / (16 * MB);
  const long long wg = (long long)blockIdx.x * 4 + wave;
  if (wg >= (long long)pixTiles * coutTiles) return;  // wave-uniform
  const int pixTile = (int)(wg / coutTiles);
  const int coutTile = (int)(wg % coutTiles);
  const int coutBase = coutTile * (16 * MB);

  int pix[2], nb[2], oy[2], ox[2];
  bool pv[2];
#pragma unroll
  for (int p = 0; p < 2; ++p) {
    int pp = pixTile * 32 + p * 16 + n16;
    pv[p] = pp < npix;
    if (!pv[p]) pp = npix - 1;
    pix[p] = pp;
    ox[p] = pp % OW;
    const int tq = pp / OW;
    oy[p] = tq % OH;
    nb[p] = tq / OH;
  }

  const _Float16* arow0 = wp + (long long)(coutBase + n16) * KTOT;
  const _Float16* arow1 = wp + (long long)(coutBase + 16 + n16) * KTOT;

  v8f acc00 = zero8(), acc01 = zero8(), acc10 = zero8(), acc11 = zero8();

  for (int ky = 0; ky < KH; ++ky) {
    for (int kx = 0; kx < KW; ++kx) {
      const _Float16* colPtr[2];
      bool valid[2];
#pragma unroll
      for (int p = 0; p < 2; ++p) {
        const int iy = oy[p] * STRIDE - PAD + ky;
        const int ix = ox[p] * STRIDE - PAD + kx;
        valid[p] = (iy >= 0) & (iy < IH) & (ix >= 0) & (ix < IW);
        colPtr[p] = in + (((long long)nb[p] * IH + iy) * IW + ix) * CIN;
      }
      const int aoff = (ky * KW + kx) * CIN;
#pragma unroll
      for (int kc = 0; kc < CIN; kc += 32) {
        const v16h b0 = load_b_frag(colPtr[0] + kc, half16, valid[0]);
        const v16h b1 = load_b_frag(colPtr[1] + kc, half16, valid[1]);
        const v16h a0 = load_a_frag(arow0 + aoff + kc, half8);
        acc00 = wmma16(a0, b0, acc00);
        acc01 = wmma16(a0, b1, acc01);
        if (MB == 2) {
          const v16h a1 = load_a_frag(arow1 + aoff + kc, half8);
          acc10 = wmma16(a1, b0, acc10);
          acc11 = wmma16(a1, b1, acc11);
        }
      }
    }
  }

  auto store_tile = [&](const v8f& acc, int cb0, int p) {
    if (!pv[p]) return;
    float f8[8];
#pragma unroll
    for (int r = 0; r < 8; ++r) {
      const int c = cb0 + half8 + r;
      float v = acc[r] + ((c < COUT_REAL) ? bias[c] : 0.f);
      if (RELU) v = v > 0.f ? v : 0.f;
      f8[r] = v;
    }
    if (WF16) {
      v8h hv;
#pragma unroll
      for (int r = 0; r < 8; ++r) hv[r] = (_Float16)f8[r];
      *reinterpret_cast<v8h*>(out16 + (long long)pix[p] * COUT + cb0 + half8) =
          hv;
    }
    if (WF32) {
#pragma unroll
      for (int r = 0; r < 8; ++r) {
        const int c = cb0 + half8 + r;
        if (c < COUT_REAL)
          out32[(((long long)nb[p] * COUT_REAL + c) * OH + oy[p]) * OW + ox[p]] =
              f8[r];
      }
    }
  };
  store_tile(acc00, coutBase, 0);
  store_tile(acc01, coutBase, 1);
  if (MB == 2) {
    store_tile(acc10, coutBase + 16, 0);
    store_tile(acc11, coutBase + 16, 1);
  }
}

// ---------------------------------------------------------------------------
// Transposed conv k4 s2 "SAME" (lhs-dilation 2, pad (2,2), unflipped kernel),
// decomposed into 4 output-parity phases (blockIdx.z = py*2+px); each phase is
// a 2x2-tap conv:  ky = 2*tY + py, iy = y + tY - (1-py)   (same for x).
// 2x2 register blocking (COUT >= 128 for all decoder layers).
// ---------------------------------------------------------------------------
template <int CIN, int COUT>
__global__ __launch_bounds__(128) void vq_convt_fwd(
    const _Float16* __restrict__ in, const _Float16* __restrict__ wp,
    const float* __restrict__ bias, _Float16* __restrict__ out16, int NB,
    int IH, int IW) {
  constexpr int KTOT = 16 * CIN;
  const int py = (int)(blockIdx.z >> 1);
  const int px = (int)(blockIdx.z & 1);
  const int lane = threadIdx.x & 31;
  const int wave = threadIdx.x >> 5;
  const int n16 = lane & 15;
  const int half8 = (lane >> 4) << 3;
  const int half16 = (lane >> 4) << 4;

  const int npix = NB * IH * IW;  // per phase
  const int pixTiles = (npix + 31) >> 5;
  const int coutTiles = COUT / 32;
  const long long wg = (long long)blockIdx.x * 4 + wave;
  if (wg >= (long long)pixTiles * coutTiles) return;  // wave-uniform
  const int pixTile = (int)(wg / coutTiles);
  const int coutTile = (int)(wg % coutTiles);
  const int coutBase = coutTile * 32;

  int nb[2], yy[2], xx[2];
  bool pv[2];
#pragma unroll
  for (int p = 0; p < 2; ++p) {
    int pp = pixTile * 32 + p * 16 + n16;
    pv[p] = pp < npix;
    if (!pv[p]) pp = npix - 1;
    xx[p] = pp % IW;
    const int tq = pp / IW;
    yy[p] = tq % IH;
    nb[p] = tq / IH;
  }

  const _Float16* arow0 = wp + (long long)(coutBase + n16) * KTOT;
  const _Float16* arow1 = wp + (long long)(coutBase + 16 + n16) * KTOT;

  v8f acc00 = zero8(), acc01 = zero8(), acc10 = zero8(), acc11 = zero8();

#pragma unroll
  for (int tY = 0; tY < 2; ++tY) {
#pragma unroll
    for (int tX = 0; tX < 2; ++tX) {
      const _Float16* colPtr[2];
      bool valid[2];
#pragma unroll
      for (int p = 0; p < 2; ++p) {
        const int iy = yy[p] + tY - (1 - py);
        const int ix = xx[p] + tX - (1 - px);
        valid[p] = (iy >= 0) & (iy < IH) & (ix >= 0) & (ix < IW);
        colPtr[p] = in + (((long long)nb[p] * IH + iy) * IW + ix) * CIN;
      }
      const int ky = 2 * tY + py;
      const int kx = 2 * tX + px;
      const int aoff = (ky * 4 + kx) * CIN;
#pragma unroll
      for (int kc = 0; kc < CIN; kc += 32) {
        const v16h b0 = load_b_frag(colPtr[0] + kc, half16, valid[0]);
        const v16h b1 = load_b_frag(colPtr[1] + kc, half16, valid[1]);
        const v16h a0 = load_a_frag(arow0 + aoff + kc, half8);
        acc00 = wmma16(a0, b0, acc00);
        acc01 = wmma16(a0, b1, acc01);
        const v16h a1 = load_a_frag(arow1 + aoff + kc, half8);
        acc10 = wmma16(a1, b0, acc10);
        acc11 = wmma16(a1, b1, acc11);
      }
    }
  }

  const int OW = IW * 2;
  auto store_tile = [&](const v8f& acc, int cb0, int p) {
    if (!pv[p]) return;
    const int oy = 2 * yy[p] + py;
    const int ox = 2 * xx[p] + px;
    const long long opix = ((long long)nb[p] * (IH * 2) + oy) * OW + ox;
    v8h hv;
#pragma unroll
    for (int r = 0; r < 8; ++r) {
      const int c = cb0 + half8 + r;
      float v = acc[r] + bias[c];
      v = v > 0.f ? v : 0.f;  // decoder layers are all ReLU
      hv[r] = (_Float16)v;
    }
    *reinterpret_cast<v8h*>(out16 + opix * COUT + cb0 + half8) = hv;
  };
  store_tile(acc00, coutBase, 0);
  store_tile(acc01, coutBase, 1);
  store_tile(acc10, coutBase + 16, 0);
  store_tile(acc11, coutBase + 16, 1);
}

// ---------------------------------------------------------------------------
// Codebook prep: f32 -> f16 copy + per-row squared norms.
// ---------------------------------------------------------------------------
__global__ void vq_cb_prep(const float* __restrict__ cb,
                           _Float16* __restrict__ cb16,
                           float* __restrict__ cbn2) {
  const int t = blockIdx.x * blockDim.x + threadIdx.x;
  if (t < 1024 * 64) cb16[t] = (_Float16)cb[t];
  if (t < 1024) {
    float s = 0.f;
    for (int c = 0; c < 64; ++c) {
      const float v = cb[t * 64 + c];
      s += v * v;
    }
    cbn2[t] = s;
  }
}

// ---------------------------------------------------------------------------
// VQ argmin: scores[pix][code] = |c|^2 - 2 * z . c  via WMMA (codes on M,
// pixels on N, K = 64).  Running min per lane, cross-half shfl reduce.
// ---------------------------------------------------------------------------
__global__ __launch_bounds__(128) void vq_argmin_k(
    const _Float16* __restrict__ z16, const _Float16* __restrict__ cb16,
    const float* __restrict__ cbn2, int* __restrict__ idx, int npix) {
  const int lane = threadIdx.x & 31;
  const int wave = threadIdx.x >> 5;
  const int n16 = lane & 15;
  const int half8 = (lane >> 4) << 3;
  const int half16 = (lane >> 4) << 4;
  const int pixTiles = (npix + 15) >> 4;
  const int tile = blockIdx.x * 4 + wave;
  if (tile >= pixTiles) return;
  int pix = tile * 16 + n16;
  const bool pv = pix < npix;
  if (!pv) pix = npix - 1;
  const _Float16* colPtr = z16 + (long long)pix * 64;
  const v16h b0 = load_b_frag(colPtr, half16, true);
  const v16h b1 = load_b_frag(colPtr + 32, half16, true);
  float bestV = 3.4e38f;
  int bestI = 0;
  for (int ct = 0; ct < 64; ++ct) {
    const _Float16* arow = cb16 + (long long)((ct << 4) + n16) * 64;
    v8f acc = zero8();
    acc = wmma16(load_a_frag(arow, half8), b0, acc);
    acc = wmma16(load_a_frag(arow + 32, half8), b1, acc);
#pragma unroll
    for (int r = 0; r < 8; ++r) {
      const int code = (ct << 4) + half8 + r;
      const float s = cbn2[code] - 2.f * acc[r];
      if (s < bestV) {
        bestV = s;
        bestI = code;
      }
    }
  }
  const float ov = __shfl_xor(bestV, 16, 32);
  const int oi = __shfl_xor(bestI, 16, 32);
  if (ov < bestV || (ov == bestV && oi < bestI)) {
    bestV = ov;
    bestI = oi;
  }
  if (pv && lane < 16) idx[pix] = bestI;
}

// ---------------------------------------------------------------------------
// Gather: z_q f32 NCHW -> d_out, and f16 NHWC -> ws (decoder input; note
// z + sg(z_q - z) == z_q in the forward pass).
// ---------------------------------------------------------------------------
__global__ void vq_gather(const int* __restrict__ idx,
                          const float* __restrict__ cb,
                          float* __restrict__ zq32, _Float16* __restrict__ zq16,
                          int NB, int H, int W) {
  const long long t = (long long)blockIdx.x * blockDim.x + threadIdx.x;
  const long long total = (long long)NB * H * W * 64;
  if (t >= total) return;
  const int c = (int)(t & 63);
  const long long pix = t >> 6;
  const int x = (int)(pix % W);
  const int y = (int)((pix / W) % H);
  const int nb = (int)(pix / ((long long)W * H));
  const float v = cb[(long long)idx[pix] * 64 + c];
  zq16[t] = (_Float16)v;
  zq32[(((long long)nb * 64 + c) * H + y) * W + x] = v;
}

// ---------------------------------------------------------------------------
// Host launcher
// ---------------------------------------------------------------------------
static inline size_t alignup(size_t v) { return (v + 255) & ~(size_t)255; }

extern "C" void kernel_launch(void* const* d_in, const int* in_sizes, int n_in,
                              void* d_out, int out_size, void* d_ws,
                              size_t ws_size, hipStream_t stream) {
  (void)in_sizes; (void)n_in; (void)out_size; (void)ws_size;
  const float* x      = (const float*)d_in[0];
  const float* enc_w0 = (const float*)d_in[1];
  const float* enc_b0 = (const float*)d_in[2];
  const float* enc_w1 = (const float*)d_in[3];
  const float* enc_b1 = (const float*)d_in[4];
  const float* enc_w2 = (const float*)d_in[5];
  const float* enc_b2 = (const float*)d_in[6];
  const float* proj_w = (const float*)d_in[7];
  const float* proj_b = (const float*)d_in[8];
  const float* dec_w0 = (const float*)d_in[9];
  const float* dec_b0 = (const float*)d_in[10];
  const float* dec_w1 = (const float*)d_in[11];
  const float* dec_b1 = (const float*)d_in[12];
  const float* dec_w2 = (const float*)d_in[13];
  const float* dec_b2 = (const float*)d_in[14];
  const float* out_w  = (const float*)d_in[15];
  const float* out_b  = (const float*)d_in[16];
  const float* cb     = (const float*)d_in[17];

  float* recon = (float*)d_out;                      // [16,3,256,256]
  float* z32   = (float*)d_out + 3145728;            // [16,64,32,32]
  float* zq32  = (float*)d_out + 4194304;            // [16,64,32,32]

  char* ws = (char*)d_ws;
  size_t off = 0;
  auto alloc = [&](size_t bytes) {
    size_t o = off;
    off = alignup(off + bytes);
    return ws + o;
  };
  _Float16* W_E0 = (_Float16*)alloc(128ull * 64 * 2);
  _Float16* W_E1 = (_Float16*)alloc(256ull * 2048 * 2);
  _Float16* W_E2 = (_Float16*)alloc(512ull * 4096 * 2);
  _Float16* W_PR = (_Float16*)alloc(64ull * 512 * 2);
  _Float16* W_D0 = (_Float16*)alloc(512ull * 1024 * 2);
  _Float16* W_D1 = (_Float16*)alloc(256ull * 8192 * 2);
  _Float16* W_D2 = (_Float16*)alloc(128ull * 4096 * 2);
  _Float16* W_OU = (_Float16*)alloc(16ull * 1152 * 2);
  _Float16* CB16 = (_Float16*)alloc(1024ull * 64 * 2);
  float*    CBN2 = (float*)alloc(1024ull * 4);
  int*      IDX  = (int*)alloc(16384ull * 4);
  char* P1 = (char*)alloc(134217728ull);  // a0 / zq16 / d1
  char* P2 = (char*)alloc(268435456ull);  // col0 / a1 / z16 / d0 / d2

  _Float16* col0 = (_Float16*)P2;
  _Float16* a0   = (_Float16*)P1;
  _Float16* a1   = (_Float16*)P2;
  _Float16* a2   = (_Float16*)alloc(16777216ull);  // own region (P1 busy)
  _Float16* z16  = (_Float16*)P2;
  _Float16* zq16 = (_Float16*)P1;
  _Float16* d0   = (_Float16*)P2;
  _Float16* d1   = (_Float16*)P1;
  _Float16* d2   = (_Float16*)P2;

  auto cdiv = [](long long a, long long b) { return (int)((a + b - 1) / b); };

  // -- weight repack + codebook prep ---------------------------------------
  vq_pack_w<<<cdiv(128ll * 64, 256), 256, 0, stream>>>(enc_w0, W_E0, 128, 3, 16, 64, 128);
  vq_pack_w<<<cdiv(256ll * 2048, 256), 256, 0, stream>>>(enc_w1, W_E1, 256, 128, 16, 2048, 256);
  vq_pack_w<<<cdiv(512ll * 4096, 256), 256, 0, stream>>>(enc_w2, W_E2, 512, 256, 16, 4096, 512);
  vq_pack_w<<<cdiv(64ll * 512, 256), 256, 0, stream>>>(proj_w, W_PR, 64, 512, 1, 512, 64);
  vq_pack_w<<<cdiv(512ll * 1024, 256), 256, 0, stream>>>(dec_w0, W_D0, 512, 64, 16, 1024, 512);
  vq_pack_w<<<cdiv(256ll * 8192, 256), 256, 0, stream>>>(dec_w1, W_D1, 256, 512, 16, 8192, 256);
  vq_pack_w<<<cdiv(128ll * 4096, 256), 256, 0, stream>>>(dec_w2, W_D2, 128, 256, 16, 4096, 128);
  vq_pack_w<<<cdiv(16ll * 1152, 256), 256, 0, stream>>>(out_w, W_OU, 3, 128, 9, 1152, 16);
  vq_cb_prep<<<cdiv(1024ll * 64, 256), 256, 0, stream>>>(cb, CB16, CBN2);

  // -- encoder --------------------------------------------------------------
  vq_im2col_e0<<<cdiv(16ll * 128 * 128 * 64, 256), 256, 0, stream>>>(x, col0, 16);
  // enc0 as GEMM over the im2col buffer (K=64, 1x1): 8192 pixTiles * 4 coutTiles
  vq_conv_fwd<64, 128, 1, 1, 1, 0, true, true, false, 128>
      <<<cdiv(8192ll * 4, 4), 128, 0, stream>>>(col0, W_E0, enc_b0, a0,
                                                nullptr, 16, 128, 128, 128, 128);
  // enc1: 2048 pixTiles * 8 coutTiles
  vq_conv_fwd<128, 256, 4, 4, 2, 1, true, true, false, 256>
      <<<cdiv(2048ll * 8, 4), 128, 0, stream>>>(a0, W_E1, enc_b1, a1, nullptr,
                                                16, 128, 128, 64, 64);
  // enc2: 512 pixTiles * 16 coutTiles
  vq_conv_fwd<256, 512, 4, 4, 2, 1, true, true, false, 512>
      <<<cdiv(512ll * 16, 4), 128, 0, stream>>>(a1, W_E2, enc_b2, a2, nullptr,
                                                16, 64, 64, 32, 32);
  // proj 1x1: 512 pixTiles * 2 coutTiles; f16 z for VQ + f32 z (NCHW) to d_out
  vq_conv_fwd<512, 64, 1, 1, 1, 0, false, true, true, 64>
      <<<cdiv(512ll * 2, 4), 128, 0, stream>>>(a2, W_PR, proj_b, z16, z32, 16,
                                               32, 32, 32, 32);

  // -- vector quantization --------------------------------------------------
  vq_argmin_k<<<cdiv(1024ll, 4), 128, 0, stream>>>(z16, CB16, CBN2, IDX, 16384);
  vq_gather<<<cdiv(16384ll * 64, 256), 256, 0, stream>>>(IDX, cb, zq32, zq16,
                                                         16, 32, 32);

  // -- decoder (phase-decomposed transposed convs) --------------------------
  {
    dim3 g(cdiv(512ll * 16, 4), 1, 4);  // 512 pixTiles * 16 coutTiles / phase
    vq_convt_fwd<64, 512><<<g, 128, 0, stream>>>(zq16, W_D0, dec_b0, d0, 16, 32, 32);
  }
  {
    dim3 g(cdiv(2048ll * 8, 4), 1, 4);
    vq_convt_fwd<512, 256><<<g, 128, 0, stream>>>(d0, W_D1, dec_b1, d1, 16, 64, 64);
  }
  {
    dim3 g(cdiv(8192ll * 4, 4), 1, 4);
    vq_convt_fwd<256, 128><<<g, 128, 0, stream>>>(d1, W_D2, dec_b2, d2, 16, 128, 128);
  }

  // -- output conv 3x3 p1 (Cout padded 3->16), f32 NCHW to d_out ------------
  // 32768 pixTiles * 1 coutTile
  vq_conv_fwd<128, 16, 3, 3, 1, 1, false, false, true, 3>
      <<<cdiv(32768ll * 1, 4), 128, 0, stream>>>(d2, W_OU, out_b, nullptr,
                                                 recon, 16, 256, 256, 256, 256);
}
